// MemristorConv1d_42494406427032
// MI455X (gfx1250) — 8-bit matrix operand order repaired and checked by reading, compiled, not run on hardware
//
#include <hip/hip_runtime.h>
#include <stdint.h>

typedef int v8i __attribute__((ext_vector_type(8)));

#define KTAPS      31
#define CCH        512
#define TT         1024
#define CUR_SCALE_F 8020.0f
#define VMAX_F      0.6f
#define QSTRIDE    1088

__global__ __launch_bounds__(256)
void memristor_conv1d_wmma(const float* __restrict__ x,
                           const float* __restrict__ w,
                           const float* __restrict__ bias,
                           const float* __restrict__ in_f_p,
                           const float* __restrict__ out_f_p,
                           float* __restrict__ out)
{
    __shared__ uint8_t lds_q[8 * QSTRIDE];

    const int tid  = (int)threadIdx.x;
    const int wave = tid >> 5;
    const int lane = tid & 31;
    const int bc   = (int)blockIdx.x * 8 + wave;
    const int b    = bc >> 9;
    const int c    = bc & 511;

    const float in_f  = in_f_p[0];
    const float out_f = out_f_p[0];
    const float bia   = bias[c];

    uint8_t* qw = lds_q + wave * QSTRIDE;

    if (lane < 16) {
        int off = (lane < 4) ? (4 * lane) : (1040 + 4 * (lane - 4));
        *(uint32_t*)(qw + off) = 0u;
    }

    const float* xc = x + ((size_t)b * CCH + c) * TT;
    #pragma unroll
    for (int g = 0; g < 8; ++g) {
        int t4 = (lane + 32 * g) * 4;
        float4 xv = *(const float4*)(xc + t4);
        int q0 = (int)rintf(fminf(fmaxf(xv.x * in_f, -1.f), 1.f) * 127.f);
        int q1 = (int)rintf(fminf(fmaxf(xv.y * in_f, -1.f), 1.f) * 127.f);
        int q2 = (int)rintf(fminf(fmaxf(xv.z * in_f, -1.f), 1.f) * 127.f);
        int q3 = (int)rintf(fminf(fmaxf(xv.w * in_f, -1.f), 1.f) * 127.f);
        uint32_t d = (uint32_t)(q0 & 0xFF)         | ((uint32_t)(q1 & 0xFF) << 8) |
                     ((uint32_t)(q2 & 0xFF) << 16) | ((uint32_t)(q3 & 0xFF) << 24);
        *(uint32_t*)(qw + 16 + t4) = d;
    }

    const int ncol = lane & 15;
    const int half = lane >> 4;

    v8i afrag = {};
    if (ncol < 9) {
        const int dlt = ncol / 3;
        const int p   = ncol - 3 * dlt;
        const float* wp = w + ((size_t)p * CCH + c) * KTAPS;
        #pragma unroll
        for (int i = 0; i < 8; ++i) {
            int kb = 32 * (i >> 2) + 4 * (i & 3) + 16 * half;
            uint32_t d = 0;
            #pragma unroll
            for (int j = 0; j < 4; ++j) {
                int k = kb + j - 16 * dlt;
                int sv = 0;
                if (k >= 0 && k < KTAPS)
                    sv = (int)rintf(wp[k] * CUR_SCALE_F);
                d |= (uint32_t)(sv & 0xFF) << (8 * j);
            }
            afrag[i] = (int)d;
        }
    }

    __syncthreads();

    const uint8_t* qbase = qw + 1 + ncol + 16 * half;
    auto loadB = [&](int t0) -> v8i {
        const uint8_t* qp = qbase + t0;
        v8i f;
        #pragma unroll
        for (int i = 0; i < 8; ++i) {
            int kb = ((i >> 2) * 32) + ((i & 3) * 4);
            uint32_t d = (uint32_t)qp[kb]
                       | ((uint32_t)qp[kb + 1] << 8)
                       | ((uint32_t)qp[kb + 2] << 16)
                       | ((uint32_t)qp[kb + 3] << 24);
            f[i] = (int)d;
        }
        return f;
    };

    const float adc_mul = VMAX_F * 256.0f / 127.0f;
    float* oc = out + ((size_t)b * CCH + c) * TT;

    v8i bfrag = loadB(0);

    for (int tile = 0; tile < 22; ++tile) {
        const int t0 = tile * 48;

        v8i acc = {};
        v8i dmat = __builtin_amdgcn_wmma_i32_16x16x64_iu8(
             true, afrag,  true, bfrag, acc,
             false,  false);

        const int t0n = (tile < 21) ? (t0 + 48) : 0;
        v8i bnext = loadB(t0n);

        float adc[8];
        #pragma unroll
        for (int r = 0; r < 8; ++r)
            adc[r] = fminf(fmaxf(rintf((float)dmat[r] * adc_mul) * 0.00390625f, -32.f), 32.f);

        float a22 = __shfl(adc[0], lane | 16, 32);

        if (half == 0) {
            float o0 = (2.0f * adc[0] + adc[1] + 0.5f * adc[2]) * out_f + bia;
            float o1 = (2.0f * adc[3] + adc[4] + 0.5f * adc[5]) * out_f + bia;
            float o2 = (2.0f * adc[6] + adc[7] + 0.5f * a22   ) * out_f + bia;
            oc[t0 + ncol] = o0;
            if (t0 + 16 + ncol < TT) oc[t0 + 16 + ncol] = o1;
            if (t0 + 32 + ncol < TT) oc[t0 + 32 + ncol] = o2;
        }

        bfrag = bnext;
    }
}

extern "C" void kernel_launch(void* const* d_in, const int* in_sizes, int n_in,
                              void* d_out, int out_size, void* d_ws, size_t ws_size,
                              hipStream_t stream) {
    (void)in_sizes; (void)n_in; (void)out_size; (void)d_ws; (void)ws_size;
    const float* x     = (const float*)d_in[0];
    const float* wbits = (const float*)d_in[1];
    const float* bias  = (const float*)d_in[2];
    const float* in_f  = (const float*)d_in[3];
    const float* out_f = (const float*)d_in[4];
    float* out = (float*)d_out;

    dim3 grid(8 * CCH / 8);
    dim3 block(256);
    memristor_conv1d_wmma<<<grid, block, 0, stream>>>(x, wbits, bias, in_f, out_f, out);
}
